// CGMLP_8040178778149
// MI455X (gfx1250) — matrix operand order repaired and checked by reading, not compiled, not run on hardware
//
#include <hip/hip_runtime.h>

#define HN 256
#define NCH 4
#define NNODE 10000
#define NEDGE 160000
#define TEM 32            // edges per block in the fused edge kernel
#define LDA 34            // ushort stride of the K-chunk stage (16x32 bf16 + pad)
#define LD1 258           // ushort stride of the 256-wide L1 stage

typedef __attribute__((ext_vector_type(16))) __bf16 bf16x16;
typedef __attribute__((ext_vector_type(8)))  float   f32x8;

// ---------- small helpers ----------
static __device__ __forceinline__ unsigned short f2bf(float f) {
  unsigned int u = __float_as_uint(f);
  u += 0x7fffu + ((u >> 16) & 1u);          // round-to-nearest-even
  return (unsigned short)(u >> 16);
}
static __device__ __forceinline__ unsigned int pack2bf(float a, float b) {
  return (unsigned int)f2bf(a) | ((unsigned int)f2bf(b) << 16);
}
static __device__ __forceinline__ float bf2f(unsigned short u) {
  return __uint_as_float(((unsigned int)u) << 16);
}
static __device__ __forceinline__ float sigm_f(float x) { return 1.0f / (1.0f + __expf(-x)); }
static __device__ __forceinline__ float silu_f(float x) { return x * sigm_f(x); }

static __device__ __forceinline__ f32x8 zero8() {
  f32x8 v;
#pragma unroll
  for (int i = 0; i < 8; ++i) v[i] = 0.0f;
  return v;
}

static __device__ __forceinline__ f32x8 wmma_bf16(bf16x16 a, bf16x16 b, f32x8 c) {
  return __builtin_amdgcn_wmma_f32_16x16x32_bf16(false, a, false, b, (short)0, c, false, false);
}

// A fragment (16x32 bf16, M x K) from LDS, rows of `ld` ushorts, kofs in ushorts (even).
// This code's own k order (not ISA table order): lane m=lane&15, half hs=lane>>4; k = {0..15}+16*hs (contiguous, matches B).
static __device__ __forceinline__ bf16x16 load_afrag(const unsigned short* sbase, int ld, int kofs) {
  const int lane = threadIdx.x & 31;
  const int m  = lane & 15;
  const int hs = lane >> 4;
  const unsigned int* rp = (const unsigned int*)(sbase + m * ld + kofs);
  union { unsigned int u[8]; bf16x16 v; } f;
#pragma unroll
  for (int r = 0; r < 4; ++r) {
    f.u[r]     = rp[hs * 8 + r];
    f.u[4 + r] = rp[hs * 8 + 4 + r];
  }
  return f.v;
}

// B fragment (32x16 bf16, K x N) straight from pre-packed global weights:
// WT is [256][kpairs] u32, u32 j packs rows k=2j,2j+1 of column n.
// lane: n = ncol + (lane&15), K half = (lane>>4)*16  -> 8 contiguous dwords.
static __device__ __forceinline__ bf16x16 load_bfrag(const unsigned int* WT, int kpairs,
                                                     int ncol, int kpbase) {
  const int lane = threadIdx.x & 31;
  const int hs = lane >> 4;
  const unsigned int* rp = WT + (size_t)(ncol + (lane & 15)) * kpairs + kpbase + hs * 8;
  union { unsigned int u[8]; bf16x16 v; } f;
#pragma unroll
  for (int q = 0; q < 8; ++q) f.u[q] = rp[q];
  return f.v;
}

// K=256 GEMM: A = 32x256 bf16 staged in LDS (stride LD1), B = packed weights (128 k-pairs).
static __device__ __forceinline__ void gemm256_lds(const unsigned short* sL, const unsigned int* WT,
                                                   f32x8 (&acc)[2][2], int wave) {
  for (int kc = 0; kc < 8; ++kc) {
    bf16x16 a0 = load_afrag(sL,            LD1, kc * 32);
    bf16x16 a1 = load_afrag(sL + 16 * LD1, LD1, kc * 32);
#pragma unroll
    for (int ntl = 0; ntl < 2; ++ntl) {
      bf16x16 b = load_bfrag(WT, 128, (wave * 2 + ntl) * 16, kc * 16);
      acc[0][ntl] = wmma_bf16(a0, b, acc[0][ntl]);
      acc[1][ntl] = wmma_bf16(a1, b, acc[1][ntl]);
    }
  }
}

// ---------- kernel: pack f32 weights [K][256] -> bf16 WT [256][kpairs] (k-pair packed) ----------
__global__ __launch_bounds__(256) void pack_wt_kernel(const float* __restrict__ W,
                                                      unsigned int* __restrict__ WT,
                                                      int K, int kpairs) {
  int idx = blockIdx.x * 256 + threadIdx.x;     // total = 256 * kpairs
  int n  = idx / kpairs;
  int kp = idx % kpairs;
  int k0 = 2 * kp;
  float a = (k0     < K) ? W[(size_t)k0 * HN + n]       : 0.0f;
  float b = (k0 + 1 < K) ? W[(size_t)(k0 + 1) * HN + n] : 0.0f;
  WT[idx] = pack2bf(a, b);
}

// ---------- kernel: coord_diff, radial, and global sum(radial^2) ----------
__global__ __launch_bounds__(256) void radial_kernel(const float* __restrict__ coord,
                                                     const int* __restrict__ ei,
                                                     float* __restrict__ cd,
                                                     float* __restrict__ rad,
                                                     float* __restrict__ nacc) {
  __shared__ float sAcc[16];
  int t = threadIdx.x;
  if (t < 16) sAcc[t] = 0.0f;
  __syncthreads();
  int e = blockIdx.x * 256 + t;                 // grid = NEDGE/256 exactly
  int r = ei[e], c = ei[NEDGE + e];
  float d[12];
#pragma unroll
  for (int i = 0; i < 12; ++i) {
    d[i] = coord[(size_t)r * 12 + i] - coord[(size_t)c * 12 + i];
    cd[(size_t)e * 12 + i] = d[i];
  }
#pragma unroll
  for (int ci = 0; ci < 4; ++ci)
#pragma unroll
    for (int di = 0; di < 4; ++di) {
      float v = d[ci*3]*d[di*3] + d[ci*3+1]*d[di*3+1] + d[ci*3+2]*d[di*3+2];
      rad[(size_t)e * 16 + ci * 4 + di] = v;
      atomicAdd(&sAcc[ci * 4 + di], v * v);
    }
  __syncthreads();
  if (t < 16) atomicAdd(&nacc[t], sAcc[t]);
}

__global__ void norm_fin_kernel(const float* __restrict__ nacc, float* __restrict__ inv) {
  int i = threadIdx.x;
  if (i < 16) {
    float n = sqrtf(nacc[i]);
    inv[i] = 1.0f / fmaxf(n, 1e-12f);
  }
}

// ---------- fused edge kernel ----------
struct EdgeArgs {
  const float *h, *h_sv, *h_se;
  const int   *ei;
  const float *radial, *inv_norm, *coord_diff;
  const unsigned int *WTer1, *WTec1, *WTer2, *WTec2, *WTcp1, *WTcp2, *WTcm1;
  const float *cm_w2;
  const float *er_b1, *er_b2, *ec_b1, *ec_b2, *cp_b1, *cp_b2, *cm_b1;
  float *agg_h, *agg_c, *cnt;
};

__global__ __launch_bounds__(256) void edge_kernel(EdgeArgs P) {
  __shared__ __align__(16) unsigned short sAch[TEM * LDA];
  __shared__ __align__(16) unsigned short sL1a[TEM * LD1];
  __shared__ __align__(16) unsigned short sL1b[TEM * LD1];
  __shared__ int   sRow[TEM], sCol[TEM];
  __shared__ float sInv[16];
  __shared__ float sCM[TEM][NCH];

  const int t    = threadIdx.x;
  const int lane = t & 31;
  const int hs   = lane >> 4;
  const int ln   = lane & 15;
  const int wave = t >> 5;
  const int e0   = blockIdx.x * TEM;

  if (t < TEM) { sRow[t] = P.ei[e0 + t]; sCol[t] = P.ei[NEDGE + e0 + t]; }
  if (t < 16)  sInv[t] = P.inv_norm[t];
  __syncthreads();

  f32x8 accA[2][2], accB[2][2];
#pragma unroll
  for (int a = 0; a < 2; ++a)
#pragma unroll
    for (int b = 0; b < 2; ++b) { accA[a][b] = zero8(); accB[a][b] = zero8(); }

  // ===== P1: er & ec layer-1, K = 5H = 1280 (40 chunks of 32) =====
  for (int kc = 0; kc < 40; ++kc) {
    const int seg = kc >> 3, off = (kc & 7) * 32;
#pragma unroll
    for (int rep = 0; rep < 2; ++rep) {
      int p = t + rep * 256;              // 512 k-pairs
      int m = p >> 4, j = p & 15;
      const float* src;
      switch (seg) {
        case 0:  src = P.h    + (size_t)sRow[m] * HN; break;
        case 1:  src = P.h    + (size_t)sCol[m] * HN; break;
        case 2:  src = P.h_sv + (size_t)sRow[m] * HN; break;
        case 3:  src = P.h_sv + (size_t)sCol[m] * HN; break;
        default: src = P.h_se + (size_t)(e0 + m) * HN; break;
      }
      float2 v = ((const float2*)(src + off))[j];
      ((unsigned int*)(sAch + m * LDA))[j] = pack2bf(v.x, v.y);
    }
    __syncthreads();
    bf16x16 a0 = load_afrag(sAch,            LDA, 0);
    bf16x16 a1 = load_afrag(sAch + 16 * LDA, LDA, 0);
#pragma unroll
    for (int ntl = 0; ntl < 2; ++ntl) {
      int ncol = (wave * 2 + ntl) * 16;
      bf16x16 bE = load_bfrag(P.WTer1, 640, ncol, kc * 16);
      accA[0][ntl] = wmma_bf16(a0, bE, accA[0][ntl]);
      accA[1][ntl] = wmma_bf16(a1, bE, accA[1][ntl]);
      bf16x16 bC = load_bfrag(P.WTec1, 640, ncol, kc * 16);
      accB[0][ntl] = wmma_bf16(a0, bC, accB[0][ntl]);
      accB[1][ntl] = wmma_bf16(a1, bC, accB[1][ntl]);
    }
    __syncthreads();
  }

  // ===== P2: bias + silu -> stage L1 activations (bf16) =====
#pragma unroll
  for (int mt = 0; mt < 2; ++mt)
#pragma unroll
    for (int ntl = 0; ntl < 2; ++ntl) {
      int n = (wave * 2 + ntl) * 16 + ln;
      float bE = P.er_b1[n], bC = P.ec_b1[n];
#pragma unroll
      for (int i = 0; i < 8; ++i) {
        int m = mt * 16 + i + hs * 8;
        sL1a[m * LD1 + n] = f2bf(silu_f(accA[mt][ntl][i] + bE));
        sL1b[m * LD1 + n] = f2bf(silu_f(accB[mt][ntl][i] + bC));
      }
    }
  __syncthreads();

  // ===== P3: layer-2 GEMMs (K=256) =====
#pragma unroll
  for (int a = 0; a < 2; ++a)
#pragma unroll
    for (int b = 0; b < 2; ++b) { accA[a][b] = zero8(); accB[a][b] = zero8(); }
  gemm256_lds(sL1a, P.WTer2, accA, wave);
  gemm256_lds(sL1b, P.WTec2, accB, wave);

  // ===== P4: edge_feat = hc*sig(row_out) + hr*sig(col_out); atomic segment-sum over row =====
#pragma unroll
  for (int mt = 0; mt < 2; ++mt)
#pragma unroll
    for (int ntl = 0; ntl < 2; ++ntl) {
      int n = (wave * 2 + ntl) * 16 + ln;
      float bE = P.er_b2[n], bC = P.ec_b2[n];
#pragma unroll
      for (int i = 0; i < 8; ++i) {
        int m = mt * 16 + i + hs * 8;
        int rI = sRow[m], cI = sCol[m];
        float hrv = P.h[(size_t)rI * HN + n];
        float hcv = P.h[(size_t)cI * HN + n];
        float ef = hcv * sigm_f(accA[mt][ntl][i] + bE) + hrv * sigm_f(accB[mt][ntl][i] + bC);
        atomicAdd(&P.agg_h[(size_t)rI * HN + n], ef);
      }
    }

  // ===== P5: cp layer-1, K = 2H + 16 -> padded 544 (17 chunks) =====
#pragma unroll
  for (int a = 0; a < 2; ++a)
#pragma unroll
    for (int b = 0; b < 2; ++b) accA[a][b] = zero8();
  for (int kc = 0; kc < 17; ++kc) {
#pragma unroll
    for (int rep = 0; rep < 2; ++rep) {
      int p = t + rep * 256;
      int m = p >> 4, j = p & 15;
      unsigned int val;
      if (kc < 8) {
        const float* s = P.h + (size_t)sRow[m] * HN + kc * 32;
        float2 v = ((const float2*)s)[j];
        val = pack2bf(v.x, v.y);
      } else if (kc < 16) {
        const float* s = P.h + (size_t)sCol[m] * HN + (kc - 8) * 32;
        float2 v = ((const float2*)s)[j];
        val = pack2bf(v.x, v.y);
      } else {
        int kl = 2 * j;
        float v0 = (kl     < 16) ? P.radial[(size_t)(e0 + m) * 16 + kl]     * sInv[kl]     : 0.0f;
        float v1 = (kl + 1 < 16) ? P.radial[(size_t)(e0 + m) * 16 + kl + 1] * sInv[kl + 1] : 0.0f;
        val = pack2bf(v0, v1);
      }
      ((unsigned int*)(sAch + m * LDA))[j] = val;
    }
    __syncthreads();
    bf16x16 a0 = load_afrag(sAch,            LDA, 0);
    bf16x16 a1 = load_afrag(sAch + 16 * LDA, LDA, 0);
#pragma unroll
    for (int ntl = 0; ntl < 2; ++ntl) {
      bf16x16 b = load_bfrag(P.WTcp1, 272, (wave * 2 + ntl) * 16, kc * 16);
      accA[0][ntl] = wmma_bf16(a0, b, accA[0][ntl]);
      accA[1][ntl] = wmma_bf16(a1, b, accA[1][ntl]);
    }
    __syncthreads();
  }
  // P6: bias + silu -> sL1a
#pragma unroll
  for (int mt = 0; mt < 2; ++mt)
#pragma unroll
    for (int ntl = 0; ntl < 2; ++ntl) {
      int n = (wave * 2 + ntl) * 16 + ln;
      float bb = P.cp_b1[n];
#pragma unroll
      for (int i = 0; i < 8; ++i) {
        int m = mt * 16 + i + hs * 8;
        sL1a[m * LD1 + n] = f2bf(silu_f(accA[mt][ntl][i] + bb));
      }
    }
  __syncthreads();

  // ===== P7: cp layer-2 -> r_out (+b2, no act) -> sL1b =====
#pragma unroll
  for (int a = 0; a < 2; ++a)
#pragma unroll
    for (int b = 0; b < 2; ++b) accA[a][b] = zero8();
  gemm256_lds(sL1a, P.WTcp2, accA, wave);
#pragma unroll
  for (int mt = 0; mt < 2; ++mt)
#pragma unroll
    for (int ntl = 0; ntl < 2; ++ntl) {
      int n = (wave * 2 + ntl) * 16 + ln;
      float bb = P.cp_b2[n];
#pragma unroll
      for (int i = 0; i < 8; ++i) {
        int m = mt * 16 + i + hs * 8;
        sL1b[m * LD1 + n] = f2bf(accA[mt][ntl][i] + bb);
      }
    }
  __syncthreads();

  // ===== P8: cm hidden = silu(r_out @ cm_w1 + b) -> sL1a =====
#pragma unroll
  for (int a = 0; a < 2; ++a)
#pragma unroll
    for (int b = 0; b < 2; ++b) accA[a][b] = zero8();
  gemm256_lds(sL1b, P.WTcm1, accA, wave);
#pragma unroll
  for (int mt = 0; mt < 2; ++mt)
#pragma unroll
    for (int ntl = 0; ntl < 2; ++ntl) {
      int n = (wave * 2 + ntl) * 16 + ln;
      float bb = P.cm_b1[n];
#pragma unroll
      for (int i = 0; i < 8; ++i) {
        int m = mt * 16 + i + hs * 8;
        sL1a[m * LD1 + n] = f2bf(silu_f(accA[mt][ntl][i] + bb));
      }
    }
  __syncthreads();

  // ===== P9: cm = cmH @ cm_w2 (256 -> 4), tiny output, VALU dots =====
  if (t < TEM * NCH) {
    int m = t >> 2, c = t & 3;
    float s = 0.0f;
    for (int k = 0; k < HN; ++k)
      s += bf2f(sL1a[m * LD1 + k]) * P.cm_w2[k * NCH + c];
    sCM[m][c] = s;
  }
  __syncthreads();

  // ===== P10: trans = coord_diff * cm, atomic segment-sum + counts =====
#pragma unroll
  for (int rep = 0; rep < 2; ++rep) {
    int idx = t + rep * 256;
    if (idx < TEM * 12) {
      int m = idx / 12, rr = idx % 12;
      float v = P.coord_diff[(size_t)(e0 + m) * 12 + rr] * sCM[m][rr / 3];
      atomicAdd(&P.agg_c[(size_t)sRow[m] * 12 + rr], v);
    }
  }
  if (t < TEM) atomicAdd(&P.cnt[sRow[t]], 1.0f);
}

// ---------- node kernel: h_new = h + mlp2([h, agg_h]); coord_new = coord + agg_c/max(cnt,1) ----------
__global__ __launch_bounds__(256) void node_kernel(const float* __restrict__ h,
                                                   const float* __restrict__ agg_h,
                                                   const unsigned int* __restrict__ WTnd1,
                                                   const unsigned int* __restrict__ WTnd2,
                                                   const float* __restrict__ nd_b1,
                                                   const float* __restrict__ nd_b2,
                                                   const float* __restrict__ coord,
                                                   const float* __restrict__ agg_c,
                                                   const float* __restrict__ cnt,
                                                   float* __restrict__ out) {
  __shared__ __align__(16) unsigned short sA[16 * LDA];
  __shared__ __align__(16) unsigned short sL1[16 * LD1];
  const int t = threadIdx.x;
  const int lane = t & 31, hs = lane >> 4, ln = lane & 15, wave = t >> 5;
  const int n0 = blockIdx.x * 16;

  f32x8 acc[2];
  acc[0] = zero8(); acc[1] = zero8();

  // GEMM1: K = 512 (h ++ agg_h)
  for (int kc = 0; kc < 16; ++kc) {
    int m = t >> 4, j = t & 15;
    const float* src = (kc < 8) ? h     + (size_t)(n0 + m) * HN + kc * 32
                                : agg_h + (size_t)(n0 + m) * HN + (kc - 8) * 32;
    float2 v = ((const float2*)src)[j];
    ((unsigned int*)(sA + m * LDA))[j] = pack2bf(v.x, v.y);
    __syncthreads();
    bf16x16 a = load_afrag(sA, LDA, 0);
#pragma unroll
    for (int ntl = 0; ntl < 2; ++ntl) {
      bf16x16 b = load_bfrag(WTnd1, 256, (wave * 2 + ntl) * 16, kc * 16);
      acc[ntl] = wmma_bf16(a, b, acc[ntl]);
    }
    __syncthreads();
  }
  // bias + silu -> sL1
#pragma unroll
  for (int ntl = 0; ntl < 2; ++ntl) {
    int n = (wave * 2 + ntl) * 16 + ln;
    float bb = nd_b1[n];
#pragma unroll
    for (int i = 0; i < 8; ++i) {
      int m = i + hs * 8;
      sL1[m * LD1 + n] = f2bf(silu_f(acc[ntl][i] + bb));
    }
  }
  __syncthreads();

  // GEMM2: K = 256
  acc[0] = zero8(); acc[1] = zero8();
  for (int kc = 0; kc < 8; ++kc) {
    bf16x16 a = load_afrag(sL1, LD1, kc * 32);
#pragma unroll
    for (int ntl = 0; ntl < 2; ++ntl) {
      bf16x16 b = load_bfrag(WTnd2, 128, (wave * 2 + ntl) * 16, kc * 16);
      acc[ntl] = wmma_bf16(a, b, acc[ntl]);
    }
  }
  // residual write
#pragma unroll
  for (int ntl = 0; ntl < 2; ++ntl) {
    int n = (wave * 2 + ntl) * 16 + ln;
    float bb = nd_b2[n];
#pragma unroll
    for (int i = 0; i < 8; ++i) {
      int node = n0 + i + hs * 8;
      out[(size_t)node * HN + n] = h[(size_t)node * HN + n] + acc[ntl][i] + bb;
    }
  }
  // coord update (16 nodes * 12 = 192 values)
  if (t < 192) {
    int m = t / 12, rr = t % 12;
    int node = n0 + m;
    float cv = fmaxf(cnt[node], 1.0f);
    out[(size_t)NNODE * HN + (size_t)node * 12 + rr] =
        coord[(size_t)node * 12 + rr] + agg_c[(size_t)node * 12 + rr] / cv;
  }
}

// ---------- host ----------
extern "C" void kernel_launch(void* const* d_in, const int* in_sizes, int n_in,
                              void* d_out, int out_size, void* d_ws, size_t ws_size,
                              hipStream_t stream) {
  (void)in_sizes; (void)n_in; (void)out_size; (void)ws_size;
  const float* h     = (const float*)d_in[0];
  const int*   ei    = (const int*)d_in[1];
  const float* coord = (const float*)d_in[2];
  const float* h_sv  = (const float*)d_in[3];
  const float* h_se  = (const float*)d_in[4];
  const float* er_w1 = (const float*)d_in[5];  const float* er_b1 = (const float*)d_in[6];
  const float* er_w2 = (const float*)d_in[7];  const float* er_b2 = (const float*)d_in[8];
  const float* ec_w1 = (const float*)d_in[9];  const float* ec_b1 = (const float*)d_in[10];
  const float* ec_w2 = (const float*)d_in[11]; const float* ec_b2 = (const float*)d_in[12];
  const float* cp_w1 = (const float*)d_in[13]; const float* cp_b1 = (const float*)d_in[14];
  const float* cp_w2 = (const float*)d_in[15]; const float* cp_b2 = (const float*)d_in[16];
  const float* cm_w1 = (const float*)d_in[17]; const float* cm_b1 = (const float*)d_in[18];
  const float* cm_w2 = (const float*)d_in[19];
  const float* nd_w1 = (const float*)d_in[20]; const float* nd_b1 = (const float*)d_in[21];
  const float* nd_w2 = (const float*)d_in[22]; const float* nd_b2 = (const float*)d_in[23];

  char* ws = (char*)d_ws;
  size_t off = 0;
  auto take = [&](size_t bytes) -> char* {
    char* p = ws + off;
    off = (off + bytes + 255) & ~(size_t)255;
    return p;
  };
  unsigned int* WTer1 = (unsigned int*)take((size_t)640 * HN * 4);
  unsigned int* WTec1 = (unsigned int*)take((size_t)640 * HN * 4);
  unsigned int* WTcp1 = (unsigned int*)take((size_t)272 * HN * 4);
  unsigned int* WTer2 = (unsigned int*)take((size_t)128 * HN * 4);
  unsigned int* WTec2 = (unsigned int*)take((size_t)128 * HN * 4);
  unsigned int* WTcp2 = (unsigned int*)take((size_t)128 * HN * 4);
  unsigned int* WTcm1 = (unsigned int*)take((size_t)128 * HN * 4);
  unsigned int* WTnd1 = (unsigned int*)take((size_t)256 * HN * 4);
  unsigned int* WTnd2 = (unsigned int*)take((size_t)128 * HN * 4);
  float* cd    = (float*)take((size_t)NEDGE * 12 * 4);
  float* rad   = (float*)take((size_t)NEDGE * 16 * 4);
  float* nacc  = (float*)take(16 * 4);
  float* invn  = (float*)take(16 * 4);
  float* agg_h = (float*)take((size_t)NNODE * HN * 4);
  float* agg_c = (float*)take((size_t)NNODE * 12 * 4);
  float* cntb  = (float*)take((size_t)NNODE * 4);

  hipMemsetAsync(nacc,  0, 16 * 4, stream);
  hipMemsetAsync(agg_h, 0, (size_t)NNODE * HN * 4, stream);
  hipMemsetAsync(agg_c, 0, (size_t)NNODE * 12 * 4, stream);
  hipMemsetAsync(cntb,  0, (size_t)NNODE * 4, stream);

  pack_wt_kernel<<<640, 256, 0, stream>>>(er_w1, WTer1, 1280, 640);
  pack_wt_kernel<<<640, 256, 0, stream>>>(ec_w1, WTec1, 1280, 640);
  pack_wt_kernel<<<272, 256, 0, stream>>>(cp_w1, WTcp1, 528, 272);
  pack_wt_kernel<<<128, 256, 0, stream>>>(er_w2, WTer2, 256, 128);
  pack_wt_kernel<<<128, 256, 0, stream>>>(ec_w2, WTec2, 256, 128);
  pack_wt_kernel<<<128, 256, 0, stream>>>(cp_w2, WTcp2, 256, 128);
  pack_wt_kernel<<<128, 256, 0, stream>>>(cm_w1, WTcm1, 256, 128);
  pack_wt_kernel<<<256, 256, 0, stream>>>(nd_w1, WTnd1, 512, 256);
  pack_wt_kernel<<<128, 256, 0, stream>>>(nd_w2, WTnd2, 256, 128);

  radial_kernel<<<NEDGE / 256, 256, 0, stream>>>(coord, ei, cd, rad, nacc);
  norm_fin_kernel<<<1, 16, 0, stream>>>(nacc, invn);

  EdgeArgs P;
  P.h = h; P.h_sv = h_sv; P.h_se = h_se; P.ei = ei;
  P.radial = rad; P.inv_norm = invn; P.coord_diff = cd;
  P.WTer1 = WTer1; P.WTec1 = WTec1; P.WTer2 = WTer2; P.WTec2 = WTec2;
  P.WTcp1 = WTcp1; P.WTcp2 = WTcp2; P.WTcm1 = WTcm1;
  P.cm_w2 = cm_w2;
  P.er_b1 = er_b1; P.er_b2 = er_b2; P.ec_b1 = ec_b1; P.ec_b2 = ec_b2;
  P.cp_b1 = cp_b1; P.cp_b2 = cp_b2; P.cm_b1 = cm_b1;
  P.agg_h = agg_h; P.agg_c = agg_c; P.cnt = cntb;
  edge_kernel<<<NEDGE / TEM, 256, 0, stream>>>(P);

  node_kernel<<<NNODE / 16, 256, 0, stream>>>(h, agg_h, WTnd1, WTnd2, nd_b1, nd_b2,
                                              coord, agg_c, cntb, (float*)d_out);
}
